// CNN_MambaClassifier_83004537963171
// MI455X (gfx1250) — matrix operand order repaired and checked by reading, not compiled, not run on hardware
//
#include <hip/hip_runtime.h>
#include <hip/hip_bf16.h>

// ---------------------------------------------------------------------------
// CNN + Mamba classifier for MI455X (gfx1250), wave32, WMMA bf16 GEMMs.
// Conv1 implicit-GEMM stages the shared im2col window (20 embedding rows,
// 10KB) into LDS via CDNA5 async-to-LDS copies; the whole K=1280 loop then
// reads B fragments from LDS (kills the 8x per-block redundant global reads).
// ---------------------------------------------------------------------------

#define B_      8
#define SEQ_    2048
#define SEQP_   (SEQ_ + 4)        // conv1 pad 2 each side
#define EMB_    256
#define CH_     128               // conv1 out channels == DM
#define LEN_    1024              // after maxpool2
#define TOK_    (B_ * LEN_)       // 8192
#define DI_     256
#define DS_     16
#define DTR_    8
#define XP_     (DTR_ + 2 * DS_)  // 40
#define KCONV_  (EMB_ * 5)        // 1280

typedef __bf16 bf16_t;
typedef __attribute__((ext_vector_type(16))) __bf16 bf16x16;
typedef __attribute__((ext_vector_type(8)))  __bf16 bf16x8;
typedef __attribute__((ext_vector_type(8)))  float  f32x8;

static __device__ __forceinline__ float bf2f(bf16_t h) { return (float)h; }
static __device__ __forceinline__ bf16_t f2bf(float f) { return (bf16_t)f; }

// ---------------------------------------------------------------------------
// Embedding gather -> padded bf16 buffer (B, SEQ+4, EMB). Rows 0,1 and
// SEQ+2,SEQ+3 of each batch are the zero pad for the k=5 conv.
// ---------------------------------------------------------------------------
__global__ __launch_bounds__(256)
void embed_pad_kernel(const int* __restrict__ ids, const float* __restrict__ emb,
                      bf16_t* __restrict__ embP) {
    int row = blockIdx.x;              // 0 .. B*SEQP-1
    int b = row / SEQP_;
    int r = row - b * SEQP_;
    int c = threadIdx.x;               // 0..255
    float v = 0.f;
    if (r >= 2 && r < SEQ_ + 2) {
        int id = ids[b * SEQ_ + (r - 2)];
        v = emb[(size_t)id * EMB_ + c];
    }
    embP[(size_t)row * EMB_ + c] = f2bf(v);
}

// conv1_w (128,256,5) -> Wpack (128, 1280) with K index = tap*256 + c (im2col order)
__global__ __launch_bounds__(256)
void pack_wconv_kernel(const float* __restrict__ w, bf16_t* __restrict__ wp) {
    int idx = blockIdx.x * 256 + threadIdx.x;
    if (idx >= CH_ * KCONV_) return;
    int o = idx / KCONV_;
    int k = idx - o * KCONV_;
    int tap = k / EMB_;
    int c   = k - tap * EMB_;
    wp[idx] = f2bf(w[((size_t)o * EMB_ + c) * 5 + tap]);
}

__global__ __launch_bounds__(256)
void f2bf_kernel(const float* __restrict__ src, bf16_t* __restrict__ dst, int n) {
    int i = blockIdx.x * 256 + threadIdx.x;
    if (i < n) dst[i] = f2bf(src[i]);
}

// ---------------------------------------------------------------------------
// Fragment helpers (wave32 WMMA bf16; contiguous k order chosen here, same for A and B):
//  A (16x32 MxK): lanes 0-15 M=lane, K kk+0..15 contig;
//                 lanes16-31 M=lane-16, K kk+16..31 contig
//  B (32x16 KxN): lane n = lane&15 col; K run kk + (lane>>4)*16 .. +15 contig.
// ---------------------------------------------------------------------------
static __device__ __forceinline__ bf16x16 load_a_frag(const bf16_t* pa) {
    bf16x8 lo = *(const bf16x8*)pa;
    bf16x8 hi = *(const bf16x8*)(pa + 8);
    return __builtin_shufflevector(lo, hi,
               0, 1, 2, 3, 4, 5, 6, 7, 8, 9, 10, 11, 12, 13, 14, 15);
}

// ---------------------------------------------------------------------------
// Conv1 implicit GEMM with async-to-LDS staging of the im2col window.
// Block = 256 thr = 8 waves; wave w -> channel tile m0 = 16w (M=128 total);
// block -> 16 positions (N) of one batch. D[ch, b*SEQ + pos], bias+ReLU, bf16.
// ---------------------------------------------------------------------------
#define CROWS_  20                 // 16 positions + 4 halo rows
#define LDSP_   272                // 256 + 16 element pitch (32B bank rotation)

__global__ __launch_bounds__(256)
void conv_gemm_wmma_kernel(const bf16_t* __restrict__ Wp,    // (128,1280)
                           const bf16_t* __restrict__ embP,  // (B,SEQP,256)
                           bf16_t* __restrict__ convout,     // (128, B*SEQ)
                           const float* __restrict__ bias) {
    __shared__ __align__(32) bf16_t sB[CROWS_ * LDSP_];

    const int tid  = threadIdx.x;
    const int lane = tid & 31;
    const int wave = tid >> 5;
    const int b    = blockIdx.z;
    const int t0   = blockIdx.x * 16;       // first position of tile
    const int m0   = wave * 16;             // channel tile

    // ---- async-stage 20 embP rows (10KB) into LDS, 16B chunks ----
    const bf16_t* gsrc = embP + ((size_t)b * SEQP_ + t0) * EMB_;
    unsigned ldsBase = (unsigned)(size_t)(&sB[0]);
    for (int i = tid; i < CROWS_ * 32; i += 256) {   // 32 x 16B chunks per row
        int r = i >> 5;
        int c = i & 31;
        unsigned laddr = ldsBase + (unsigned)(r * (LDSP_ * 2) + c * 16);
        unsigned long long gaddr =
            (unsigned long long)(size_t)gsrc + (size_t)r * (EMB_ * 2) + (size_t)c * 16;
        asm volatile("global_load_async_to_lds_b128 %0, %1, off"
                     :: "v"(laddr), "v"(gaddr) : "memory");
    }
    asm volatile("s_wait_asynccnt 0x0" ::: "memory");
    __syncthreads();

    const int hi = lane >> 4;
    const bf16_t* pa = Wp + (size_t)(m0 + (lane & 15)) * KCONV_ + (hi ? 16 : 0);
    const int nB  = lane & 15;
    const int khB = hi ? 16 : 0;

    f32x8 acc = {};
#pragma unroll 4
    for (int kk = 0; kk < KCONV_; kk += 32) {
        bf16x16 a = load_a_frag(pa + kk);
        int k0 = kk + khB;                            // run start, mult of 16
        const bf16_t* pb = &sB[(nB + (k0 >> 8)) * LDSP_ + (k0 & 255)];
        bf16x16 bf = *(const bf16x16*)pb;             // ds_load from LDS
        acc = __builtin_amdgcn_wmma_f32_16x16x32_bf16(
                  false, a, false, bf, (short)0, acc, false, false);
    }

    // epilogue: bias + ReLU, store convout[ch * 16384 + b*2048 + t0 + n]
    const int n = t0 + nB;
    const int mbase = m0 + (hi ? 8 : 0);
    const size_t obase = (size_t)b * SEQ_ + n;
#pragma unroll
    for (int r = 0; r < 8; ++r) {
        int m = mbase + r;
        float v = acc[r] + bias[m];
        v = v > 0.f ? v : 0.f;
        convout[(size_t)m * (B_ * SEQ_) + obase] = f2bf(v);
    }
}

// ---------------------------------------------------------------------------
// Generic wave32 WMMA GEMM:   D[M,N] = X[M,K] * Wt[N,K]^T   (direct global)
//   MODE 0: store f32   MODE 1: store bf16
// ---------------------------------------------------------------------------
template <int MODE>
__global__ __launch_bounds__(256)
void gemm_wmma_kernel(const bf16_t* __restrict__ X, int lda,
                      const bf16_t* __restrict__ Wt, int ldb,
                      void* __restrict__ Out, int ldc,
                      int Nvalid, int K) {
    const int lane = threadIdx.x & 31;
    const int wave = threadIdx.x >> 5;
    const int m0 = blockIdx.y * 128 + wave * 16;
    const int n0 = blockIdx.x * 16;

    const int hi = lane >> 4;
    const int ar = m0 + (lane & 15);
    int       br = n0 + (lane & 15);
    if (br >= Nvalid) br = Nvalid - 1;        // clamp (padded N tiles stay in-bounds)

    const bf16_t* pa = X  + (size_t)ar * lda + (hi ? 16 : 0);
    const bf16_t* pb = Wt + (size_t)br * ldb + (hi ? 16 : 0);

    f32x8 acc = {};
#pragma unroll 4
    for (int kk = 0; kk < K; kk += 32) {
        bf16x16 a = load_a_frag(pa + kk);
        bf16x16 b = *(const bf16x16*)(pb + kk);
        acc = __builtin_amdgcn_wmma_f32_16x16x32_bf16(
                  false, a, false, b, (short)0, acc, false, false);
    }

    const int n = n0 + (lane & 15);
    if (n >= Nvalid) return;
    const int mbase = m0 + (hi ? 8 : 0);
#pragma unroll
    for (int r = 0; r < 8; ++r) {
        size_t oi = (size_t)(mbase + r) * ldc + n;
        if (MODE == 0) ((float*)Out)[oi] = acc[r];
        else           ((bf16_t*)Out)[oi] = f2bf(acc[r]);
    }
}

// convout (CH, B*SEQ) -> maxpool pairs -> xm (token, CH) bf16
__global__ __launch_bounds__(128)
void maxpool_tr_kernel(const bf16_t* __restrict__ convout, bf16_t* __restrict__ xm) {
    int token = blockIdx.x;                 // 0..8191
    int ch = threadIdx.x;                   // 0..127
    int b = token >> 10, l = token & 1023;
    size_t gp = (size_t)ch * (B_ * SEQ_) + (b << 11) + 2 * l;
    float a = bf2f(convout[gp]);
    float c = bf2f(convout[gp + 1]);
    xm[(size_t)token * CH_ + ch] = f2bf(a > c ? a : c);
}

// Depthwise causal conv (DCONV=4) + SiLU on xs half of xz; writes f32 + bf16
__global__ __launch_bounds__(256)
void dwconv_silu_kernel(const bf16_t* __restrict__ xz, const float* __restrict__ w,
                        const float* __restrict__ bias,
                        float* __restrict__ xs_f, bf16_t* __restrict__ xs_h) {
    int idx = blockIdx.x * 256 + threadIdx.x;   // token*256 + d
    int token = idx >> 8;
    int d = idx & 255;
    int b = token >> 10, l = token & 1023;
    float acc = bias[d];
#pragma unroll
    for (int t = 0; t < 4; ++t) {
        int li = l - 3 + t;
        if (li >= 0)
            acc += w[d * 4 + t] * bf2f(xz[((size_t)((b << 10) | li)) * 512 + d]);
    }
    float s = acc / (1.f + __expf(-acc));       // SiLU
    xs_f[idx] = s;
    xs_h[idx] = f2bf(s);
}

// dt = softplus(dtr @ dt_proj_w^T + b) ; K=8 -> plain VALU
__global__ __launch_bounds__(256)
void dtproj_kernel(const float* __restrict__ xdbl, const float* __restrict__ w,
                   const float* __restrict__ bias, float* __restrict__ dtv) {
    int idx = blockIdx.x * 256 + threadIdx.x;
    int token = idx >> 8;
    int d = idx & 255;
    const float* xr = xdbl + (size_t)token * XP_;
    float s = bias[d];
#pragma unroll
    for (int r = 0; r < DTR_; ++r) s += xr[r] * w[d * DTR_ + r];
    dtv[idx] = (s > 20.f) ? s : log1pf(__expf(s));
}

// Selective scan. Block = 256 thr handles (batch, 16 d's) x 16 states.
__global__ __launch_bounds__(256)
void scan_kernel(const float* __restrict__ dtv, const float* __restrict__ xs_f,
                 const float* __restrict__ xdbl, const bf16_t* __restrict__ xz,
                 const float* __restrict__ A_log, const float* __restrict__ Dp,
                 bf16_t* __restrict__ y_h) {
    int b    = blockIdx.x >> 4;
    int dblk = blockIdx.x & 15;
    int tid = threadIdx.x;
    int dl = tid >> 4, n = tid & 15;
    int d = dblk * 16 + dl;
    float A  = -__expf(A_log[d * DS_ + n]);
    float Dd = Dp[d];
    float h = 0.f;
    int tok0 = b << 10;
    for (int l = 0; l < LEN_; ++l) {
        int token = tok0 + l;
        float dt = dtv[(size_t)token * DI_ + d];
        float xv = xs_f[(size_t)token * DI_ + d];
        float Bv = xdbl[(size_t)token * XP_ + DTR_ + n];
        float Cv = xdbl[(size_t)token * XP_ + DTR_ + DS_ + n];
        h = __expf(dt * A) * h + (dt * xv) * Bv;
        float p = h * Cv;
        p += __shfl_xor(p, 1, 32);
        p += __shfl_xor(p, 2, 32);
        p += __shfl_xor(p, 4, 32);
        p += __shfl_xor(p, 8, 32);
        if (n == 0) {
            float zv = bf2f(xz[(size_t)token * 512 + 256 + d]);
            float yv = (p + xv * Dd) * (zv / (1.f + __expf(-zv)));
            y_h[(size_t)token * DI_ + d] = f2bf(yv);
        }
    }
}

// mean over L then FC head -> (B, 10) f32
__global__ __launch_bounds__(128)
void head_kernel(const float* __restrict__ yout, const float* __restrict__ fc_w,
                 const float* __restrict__ fc_b, float* __restrict__ out) {
    __shared__ float mean[CH_];
    int b = blockIdx.x;
    int t = threadIdx.x;                    // 0..127
    float s = 0.f;
    for (int l = 0; l < LEN_; ++l) s += yout[((size_t)((b << 10) + l)) * CH_ + t];
    mean[t] = s * (1.f / (float)LEN_);
    __syncthreads();
    if (t < 10) {
        float o = fc_b[t];
#pragma unroll 4
        for (int k = 0; k < CH_; ++k) o += mean[k] * fc_w[t * CH_ + k];
        out[b * 10 + t] = o;
    }
}

// ---------------------------------------------------------------------------
extern "C" void kernel_launch(void* const* d_in, const int* in_sizes, int n_in,
                              void* d_out, int out_size, void* d_ws, size_t ws_size,
                              hipStream_t stream) {
    (void)in_sizes; (void)n_in; (void)out_size; (void)ws_size;
    const int*   ids       = (const int*)d_in[0];
    const float* emb       = (const float*)d_in[1];
    const float* conv1_w   = (const float*)d_in[2];
    const float* conv1_b   = (const float*)d_in[3];
    const float* in_proj_w = (const float*)d_in[4];
    const float* convd_w   = (const float*)d_in[5];
    const float* convd_b   = (const float*)d_in[6];
    const float* x_proj_w  = (const float*)d_in[7];
    const float* dt_proj_w = (const float*)d_in[8];
    const float* dt_proj_b = (const float*)d_in[9];
    const float* A_log     = (const float*)d_in[10];
    const float* Dp        = (const float*)d_in[11];
    const float* out_proj_w= (const float*)d_in[12];
    const float* fc_w      = (const float*)d_in[13];
    const float* fc_b      = (const float*)d_in[14];
    float* out = (float*)d_out;

    char* wp = (char*)d_ws;
    auto carve = [&](size_t bytes) -> void* {
        void* p = (void*)wp;
        wp += (bytes + 255) & ~(size_t)255;
        return p;
    };
    bf16_t* embP    = (bf16_t*)carve((size_t)B_ * SEQP_ * EMB_ * 2);
    bf16_t* wconv_h = (bf16_t*)carve((size_t)CH_ * KCONV_ * 2);
    bf16_t* win_h   = (bf16_t*)carve((size_t)512 * 128 * 2);
    bf16_t* wxp_h   = (bf16_t*)carve((size_t)XP_ * DI_ * 2);
    bf16_t* wout_h  = (bf16_t*)carve((size_t)CH_ * DI_ * 2);
    bf16_t* convout = (bf16_t*)carve((size_t)CH_ * B_ * SEQ_ * 2);
    bf16_t* xm      = (bf16_t*)carve((size_t)TOK_ * CH_ * 2);
    bf16_t* xz      = (bf16_t*)carve((size_t)TOK_ * 512 * 2);
    float*  xs_f    = (float*) carve((size_t)TOK_ * DI_ * 4);
    bf16_t* xs_h    = (bf16_t*)carve((size_t)TOK_ * DI_ * 2);
    float*  xdbl    = (float*) carve((size_t)TOK_ * XP_ * 4);
    float*  dtv     = (float*) carve((size_t)TOK_ * DI_ * 4);
    bf16_t* y_h     = (bf16_t*)carve((size_t)TOK_ * DI_ * 2);
    float*  yout    = (float*) carve((size_t)TOK_ * CH_ * 4);

    // 1) embedding gather + pad (bf16)
    embed_pad_kernel<<<B_ * SEQP_, 256, 0, stream>>>(ids, emb, embP);
    // 2) weight packing / conversions
    pack_wconv_kernel<<<(CH_ * KCONV_ + 255) / 256, 256, 0, stream>>>(conv1_w, wconv_h);
    f2bf_kernel<<<(512 * 128 + 255) / 256, 256, 0, stream>>>(in_proj_w, win_h, 512 * 128);
    f2bf_kernel<<<(XP_ * DI_ + 255) / 256, 256, 0, stream>>>(x_proj_w, wxp_h, XP_ * DI_);
    f2bf_kernel<<<(CH_ * DI_ + 255) / 256, 256, 0, stream>>>(out_proj_w, wout_h, CH_ * DI_);

    // 3) conv1 implicit GEMM with async-LDS im2col staging
    conv_gemm_wmma_kernel<<<dim3(SEQ_ / 16, 1, B_), 256, 0, stream>>>(
        wconv_h, embP, convout, conv1_b);

    // 4) maxpool(2) + transpose -> xm (token, 128) bf16
    maxpool_tr_kernel<<<TOK_, 128, 0, stream>>>(convout, xm);

    // 5) in_proj: xz (token, 512) = xm (token,128) . in_proj_w(512,128)^T
    gemm_wmma_kernel<1><<<dim3(512 / 16, TOK_ / 128, 1), 256, 0, stream>>>(
        xm, CH_, win_h, CH_, xz, 512, 512, CH_);

    // 6) depthwise causal conv + SiLU on xs half
    dwconv_silu_kernel<<<(TOK_ * DI_) / 256, 256, 0, stream>>>(
        xz, convd_w, convd_b, xs_f, xs_h);

    // 7) x_proj: xdbl (token, 40) = xs (token,256) . x_proj_w(40,256)^T
    gemm_wmma_kernel<0><<<dim3((XP_ + 15) / 16, TOK_ / 128, 1), 256, 0, stream>>>(
        xs_h, DI_, wxp_h, DI_, xdbl, XP_, XP_, DI_);

    // 8) dt_proj + softplus
    dtproj_kernel<<<(TOK_ * DI_) / 256, 256, 0, stream>>>(xdbl, dt_proj_w, dt_proj_b, dtv);

    // 9) selective scan + D-skip + SiLU(z) gate -> y_h (token,256) bf16
    scan_kernel<<<B_ * (DI_ / 16), 256, 0, stream>>>(dtv, xs_f, xdbl, xz, A_log, Dp, y_h);

    // 10) out_proj: yout (token,128) = y (token,256) . out_proj_w(128,256)^T
    gemm_wmma_kernel<0><<<dim3(CH_ / 16, TOK_ / 128, 1), 256, 0, stream>>>(
        y_h, DI_, wout_h, DI_, yout, CH_, CH_, DI_);

    // 11) mean over L + FC head
    head_kernel<<<B_, 128, 0, stream>>>(yout, fc_w, fc_b, out);
}
